// GroupedMLP_46007689675057
// MI455X (gfx1250) — matrix operand order repaired and checked by reading, not compiled, not run on hardware
//
#include <hip/hip_runtime.h>
#include <hip/hip_bf16.h>

// ---------------------------------------------------------------------------
// Grouped MoE MLP (gate/up + SwiGLU + down) for MI455X (gfx1250).
// bf16 WMMA (v_wmma_f32_16x16x32_bf16) with fp32 accumulation. fp32 operands
// are converted with packed HW converts (v_cvt_pk_bf16_f32) while staging
// through double-buffered LDS; global loads for stage s+1 are issued before
// the WMMAs of stage s (one barrier per stage). Kernel 2's A-tile (raw bf16)
// uses the Tensor Data Mover (tensor_load_to_lds + s_wait_tensorcnt) when
// the builtin exists. All staging uses compile-time-constant register
// indices (no dynamic indexing -> no scratch spills).
// ---------------------------------------------------------------------------

#define E_ 8
#define H_ 1024
#define I_ 2816
#define T_ 8192

#define BM 128           // token rows per block tile
#define BN 128           // output cols per block tile
#define BK 32            // K per stage == one bf16 WMMA K-step
#define LD 40            // padded LDS stride (bf16 elems): 80B, 16B-aligned

typedef __attribute__((ext_vector_type(16))) __bf16 v16bf;
typedef __attribute__((ext_vector_type(2)))  __bf16 v2bf;
typedef __attribute__((ext_vector_type(8)))  float  v8f;

union Frag { v16bf v; uint4 q[2]; };

__device__ __forceinline__ unsigned short f2bf(float f) {
  union { __bf16 b; unsigned short u; } c;
  c.b = (__bf16)f;
  return c.u;
}
// one packed convert for a (lo,hi) pair
__device__ __forceinline__ unsigned pkbf(float lo, float hi) {
#if __has_builtin(__builtin_amdgcn_cvt_pk_bf16_f32)
  auto t = __builtin_amdgcn_cvt_pk_bf16_f32(lo, hi);
  unsigned u;
  __builtin_memcpy(&u, &t, 4);
  return u;
#else
  union { v2bf v; unsigned u; } c;
  c.v.x = (__bf16)lo;
  c.v.y = (__bf16)hi;
  return c.u;
#endif
}

__device__ __forceinline__ v8f vzero() {
  v8f z = {0.f, 0.f, 0.f, 0.f, 0.f, 0.f, 0.f, 0.f};
  return z;
}

__device__ __forceinline__ v8f wmma_bf16(const Frag& a, const Frag& b, v8f c) {
  return __builtin_amdgcn_wmma_f32_16x16x32_bf16(
      false, a.v, false, b.v, (short)0, c, false, false);
}

// ---------------------------------------------------------------------------
// Tensor Data Mover: 2D bf16 tile global->LDS with LDS row padding (D# per
// CDNA5 ISA). 5-arg (clang-22) vs 6-arg (clang-23).
// ---------------------------------------------------------------------------
#if __has_builtin(__builtin_amdgcn_tensor_load_to_lds)
#define HAVE_TDM 1
typedef __attribute__((ext_vector_type(4))) unsigned tdm_u32x4;
typedef __attribute__((ext_vector_type(8))) int      tdm_i32x8;
typedef __attribute__((ext_vector_type(4))) int      tdm_i32x4;

__device__ __forceinline__ void tdm_load_tile_bf16(
    unsigned lds_byte_addr, const unsigned short* gptr,
    unsigned rows_valid, unsigned row_stride_elems) {
  unsigned long long ga = (unsigned long long)(uintptr_t)gptr;

  tdm_u32x4 g0;
  g0[0] = 1u;                                   // count=1, no gather/restore
  g0[1] = lds_byte_addr;                        // lds_addr
  g0[2] = (unsigned)(ga & 0xFFFFFFFFu);
  g0[3] = (unsigned)((ga >> 32) & 0x01FFFFFFu) | (2u << 30);  // type=2

  unsigned dw0 = (1u << 16)                     // data_size = 2 bytes
               | (1u << 20)                     // pad_enable
               | (3u << 22)                     // pad_interval: 16 DWORDs
               | (3u << 25);                    // pad_amount:    4 DWORDs
  unsigned td0 = (unsigned)BK;                  // tensor_dim0
  unsigned td1 = rows_valid;                    // tensor_dim1 (OOB rows -> 0)
  unsigned dw1 = (td0 & 0xFFFFu) << 16;
  unsigned dw2 = (td0 >> 16) | ((td1 & 0xFFFFu) << 16);
  unsigned dw3 = (td1 >> 16) | ((unsigned)BK << 16);   // tile_dim0 = 32
  unsigned dw4 = (unsigned)BM;                         // tile_dim1 = 128
  unsigned dw5 = row_stride_elems;                     // tensor_dim0_stride
  tdm_i32x8 g1;
  g1[0] = (int)dw0; g1[1] = (int)dw1; g1[2] = (int)dw2; g1[3] = (int)dw3;
  g1[4] = (int)dw4; g1[5] = (int)dw5; g1[6] = 0;       g1[7] = 0;

  tdm_i32x4 gz = {0, 0, 0, 0};
#if defined(__clang_major__) && (__clang_major__ >= 23)
  tdm_i32x8 gz8 = {0, 0, 0, 0, 0, 0, 0, 0};
  __builtin_amdgcn_tensor_load_to_lds(g0, g1, gz, gz, gz8, 0);
#else
  __builtin_amdgcn_tensor_load_to_lds(g0, g1, gz, gz, 0);
#endif
}
#endif

// ---------------------------------------------------------------------------
// Kernel 1: o1 = X@gate, o2 = X@up, inter = silu(o1)*o2   (bf16 out to ws)
// grid = (I/BN, T/BM, E), block = 256 (8 waves, 2x4 wave grid)
// ---------------------------------------------------------------------------
__global__ __launch_bounds__(256)
void moe_gateup_swiglu(const float* __restrict__ x,
                       const float* __restrict__ gate,
                       const float* __restrict__ up,
                       const int* __restrict__ tpe,
                       unsigned short* __restrict__ inter) {
  const int e  = blockIdx.z;
  const int mt = blockIdx.y;
  const int nt = blockIdx.x;

  long tok_start = 0;
  for (int i = 0; i < e; ++i) tok_start += tpe[i];
  const int cnt = tpe[e];
  if (mt * BM >= cnt) return;

  __shared__ __align__(16) unsigned short Al[2][BM * LD];
  __shared__ __align__(16) unsigned short Bg[2][BN * LD];
  __shared__ __align__(16) unsigned short Bu[2][BN * LD];

  const int tid  = threadIdx.x;
  const int lane = tid & 31;
  const int wave = tid >> 5;
  const int wm   = wave >> 2;        // 0..1 : 64 rows each
  const int wn   = wave & 3;         // 0..3 : 32 cols each
  const int l16  = lane & 15;
  const int hi   = lane >> 4;

  const float* gbase = gate + (size_t)e * H_ * I_ + (size_t)nt * BN;
  const float* ubase = up   + (size_t)e * H_ * I_ + (size_t)nt * BN;
  const float* abase = x    + (size_t)tok_start * H_;

  // per-thread staging coordinates for the B tiles: one k-quad x 4 columns
  const int bkq = tid >> 5;          // 0..7  -> k = 4*bkq (wave-uniform)
  const int bnq = tid & 31;          // float4 column group

  v8f accg[4][2], accu[4][2];
#pragma unroll
  for (int i = 0; i < 4; ++i)
#pragma unroll
    for (int j = 0; j < 2; ++j) { accg[i][j] = vzero(); accu[i][j] = vzero(); }

  // staging registers for one stage (constant-indexed only)
  float4 ra[4];            // A: 4 float4/thread
  float4 rg[4], ru[4];     // B: rows k..k+3 of one float4 column group

  auto k1_load = [&](int s) {
#pragma unroll
    for (int i = 0; i < 4; ++i) {
      int idx = tid + i * 256;              // 0..1023
      int row = idx >> 3, kq = idx & 7;
      int grow = mt * BM + row;
      ra[i] = make_float4(0.f, 0.f, 0.f, 0.f);
      if (grow < cnt)
        ra[i] = *(const float4*)(abase + (size_t)grow * H_ + s * BK + kq * 4);
    }
    const float* g0 = gbase + (size_t)(s * BK + bkq * 4) * I_ + bnq * 4;
    const float* u0 = ubase + (size_t)(s * BK + bkq * 4) * I_ + bnq * 4;
#pragma unroll
    for (int r = 0; r < 4; ++r) {
      rg[r] = *(const float4*)(g0 + (size_t)r * I_);
      ru[r] = *(const float4*)(u0 + (size_t)r * I_);
    }
    // prefetch one stage further ahead (global_prefetch_b8)
    if (s + 1 < H_ / BK) {
      __builtin_prefetch(g0 + (size_t)BK * I_, 0, 3);
      __builtin_prefetch(u0 + (size_t)BK * I_, 0, 3);
    }
  };

  auto k1_store = [&](int b) {
#pragma unroll
    for (int i = 0; i < 4; ++i) {
      int idx = tid + i * 256;
      int row = idx >> 3, kq = idx & 7;
      *(uint2*)&Al[b][row * LD + kq * 4] =
          make_uint2(pkbf(ra[i].x, ra[i].y), pkbf(ra[i].z, ra[i].w));
    }
    int k = bkq * 4;
#pragma unroll
    for (int j = 0; j < 4; ++j) {          // j is compile-time after unroll
      *(uint2*)&Bg[b][(bnq * 4 + j) * LD + k] =
          make_uint2(pkbf((&rg[0].x)[j], (&rg[1].x)[j]),
                     pkbf((&rg[2].x)[j], (&rg[3].x)[j]));
      *(uint2*)&Bu[b][(bnq * 4 + j) * LD + k] =
          make_uint2(pkbf((&ru[0].x)[j], (&ru[1].x)[j]),
                     pkbf((&ru[2].x)[j], (&ru[3].x)[j]));
    }
  };

  auto k1_compute = [&](int b) {
    Frag af[4];
#pragma unroll
    for (int mf = 0; mf < 4; ++mf) {
      int r  = wm * 64 + mf * 16 + l16;
      int kb = hi * 8;
      af[mf].q[0] = *(const uint4*)&Al[b][r * LD + kb];
      af[mf].q[1] = *(const uint4*)&Al[b][r * LD + kb + 16];
    }
#pragma unroll
    for (int nf = 0; nf < 2; ++nf) {
      int n  = wn * 32 + nf * 16 + l16;
      int kb = hi * 8;
      Frag bg, bu;
      bg.q[0] = *(const uint4*)&Bg[b][n * LD + kb];
      bg.q[1] = *(const uint4*)&Bg[b][n * LD + kb + 16];
      bu.q[0] = *(const uint4*)&Bu[b][n * LD + kb];
      bu.q[1] = *(const uint4*)&Bu[b][n * LD + kb + 16];
#pragma unroll
      for (int mf = 0; mf < 4; ++mf) {
        accg[mf][nf] = wmma_bf16(af[mf], bg, accg[mf][nf]);
        accu[mf][nf] = wmma_bf16(af[mf], bu, accu[mf][nf]);
      }
    }
  };

  const int KST = H_ / BK;   // 32
  k1_load(0);
  k1_store(0);
  __syncthreads();
  for (int s = 0; s < KST; ++s) {
    int cur = s & 1, nxt = cur ^ 1;
    if (s + 1 < KST) k1_load(s + 1);   // issue globals; overlap with WMMAs
    k1_compute(cur);
    if (s + 1 < KST) k1_store(nxt);    // waits loads, converts, fills buffer
    __syncthreads();
  }

  // ---- SwiGLU epilogue, bf16 store to workspace
#pragma unroll
  for (int mf = 0; mf < 4; ++mf) {
#pragma unroll
    for (int nf = 0; nf < 2; ++nf) {
      v8f g = accg[mf][nf];
      v8f u = accu[mf][nf];
      int col = nt * BN + wn * 32 + nf * 16 + l16;
#pragma unroll
      for (int v = 0; v < 8; ++v) {
        int lrow = mt * BM + wm * 64 + mf * 16 + v + hi * 8;
        if (lrow < cnt) {
          float o1  = g[v];
          float val = (o1 / (1.f + __expf(-o1))) * u[v];
          inter[(size_t)(tok_start + lrow) * I_ + col] = f2bf(val);
        }
      }
    }
  }
}

// ---------------------------------------------------------------------------
// Kernel 2: out = inter @ down   (bf16 A from ws, fp32 out)
// grid = (H/BN, T/BM, E), block = 256
// ---------------------------------------------------------------------------
__global__ __launch_bounds__(256)
void moe_down(const unsigned short* __restrict__ inter,
              const float* __restrict__ down,
              const int* __restrict__ tpe,
              float* __restrict__ out) {
  const int e  = blockIdx.z;
  const int mt = blockIdx.y;
  const int nt = blockIdx.x;

  long tok_start = 0;
  for (int i = 0; i < e; ++i) tok_start += tpe[i];
  const int cnt = tpe[e];
  if (mt * BM >= cnt) return;

  __shared__ __align__(16) unsigned short Al[2][BM * LD];
  __shared__ __align__(16) unsigned short Bd[2][BN * LD];

  const int tid  = threadIdx.x;
  const int lane = tid & 31;
  const int wave = tid >> 5;
  const int wm   = wave >> 2;
  const int wn   = wave & 3;
  const int l16  = lane & 15;
  const int hi   = lane >> 4;

  const float* dbase = down + (size_t)e * I_ * H_ + (size_t)nt * BN;
  const unsigned short* abase = inter + (size_t)tok_start * I_;

  const int bkq = tid >> 5;
  const int bnq = tid & 31;

  v8f acc[4][2];
#pragma unroll
  for (int i = 0; i < 4; ++i)
#pragma unroll
    for (int j = 0; j < 2; ++j) acc[i][j] = vzero();

  unsigned rows_valid = (unsigned)(cnt - mt * BM);
  if (rows_valid > BM) rows_valid = BM;

  float4 rd[4];                  // B staging: rows k..k+3, one column group
#ifndef HAVE_TDM
  uint4 ua[2];                   // A staging (fallback path)
#endif

  auto k2_loadB = [&](int s) {
    const float* d0 = dbase + (size_t)(s * BK + bkq * 4) * H_ + bnq * 4;
#pragma unroll
    for (int r = 0; r < 4; ++r)
      rd[r] = *(const float4*)(d0 + (size_t)r * H_);
    if (s + 1 < I_ / BK)
      __builtin_prefetch(d0 + (size_t)BK * H_, 0, 3);
  };
  auto k2_storeB = [&](int b) {
    int k = bkq * 4;
#pragma unroll
    for (int j = 0; j < 4; ++j) {
      *(uint2*)&Bd[b][(bnq * 4 + j) * LD + k] =
          make_uint2(pkbf((&rd[0].x)[j], (&rd[1].x)[j]),
                     pkbf((&rd[2].x)[j], (&rd[3].x)[j]));
    }
  };
#ifdef HAVE_TDM
  auto k2_loadA_tdm = [&](int b, int s) {
    if (wave == 0)
      tdm_load_tile_bf16((unsigned)(uintptr_t)&Al[b][0],
                         abase + (size_t)(mt * BM) * I_ + s * BK,
                         rows_valid, (unsigned)I_);
  };
#else
  auto k2_loadA = [&](int s) {
#pragma unroll
    for (int i = 0; i < 2; ++i) {
      int idx = tid + i * 256;
      int row = idx >> 2, kq = idx & 3;
      int grow = mt * BM + row;
      ua[i] = make_uint4(0u, 0u, 0u, 0u);
      if (grow < cnt)
        ua[i] = *(const uint4*)(abase + (size_t)grow * I_ + s * BK + kq * 8);
    }
  };
  auto k2_storeA = [&](int b) {
#pragma unroll
    for (int i = 0; i < 2; ++i) {
      int idx = tid + i * 256;
      int row = idx >> 2, kq = idx & 3;
      *(uint4*)&Al[b][row * LD + kq * 8] = ua[i];
    }
  };
#endif

  auto k2_compute = [&](int b) {
    Frag af[4];
#pragma unroll
    for (int mf = 0; mf < 4; ++mf) {
      int r  = wm * 64 + mf * 16 + l16;
      int kb = hi * 8;
      af[mf].q[0] = *(const uint4*)&Al[b][r * LD + kb];
      af[mf].q[1] = *(const uint4*)&Al[b][r * LD + kb + 16];
    }
#pragma unroll
    for (int nf = 0; nf < 2; ++nf) {
      int n  = wn * 32 + nf * 16 + l16;
      int kb = hi * 8;
      Frag bd;
      bd.q[0] = *(const uint4*)&Bd[b][n * LD + kb];
      bd.q[1] = *(const uint4*)&Bd[b][n * LD + kb + 16];
#pragma unroll
      for (int mf = 0; mf < 4; ++mf)
        acc[mf][nf] = wmma_bf16(af[mf], bd, acc[mf][nf]);
    }
  };

  const int KST = I_ / BK;   // 88
#ifdef HAVE_TDM
  k2_loadA_tdm(0, 0);
#else
  k2_loadA(0);
#endif
  k2_loadB(0);
#ifndef HAVE_TDM
  k2_storeA(0);
#endif
  k2_storeB(0);
#ifdef HAVE_TDM
  __builtin_amdgcn_s_wait_tensorcnt(0);   // SALU; trivially 0 for waves 1..7
#endif
  __syncthreads();

  for (int s = 0; s < KST; ++s) {
    int cur = s & 1, nxt = cur ^ 1;
    if (s + 1 < KST) {
#ifdef HAVE_TDM
      k2_loadA_tdm(nxt, s + 1);   // TDM fills other buffer during WMMAs
#else
      k2_loadA(s + 1);
#endif
      k2_loadB(s + 1);
    }
    k2_compute(cur);
    if (s + 1 < KST) {
#ifndef HAVE_TDM
      k2_storeA(nxt);
#endif
      k2_storeB(nxt);
    }
#ifdef HAVE_TDM
    __builtin_amdgcn_s_wait_tensorcnt(0);
#endif
    __syncthreads();
  }

  // ---- fp32 epilogue store (C layout: VGPR v -> row v / v+8, lane -> col)
#pragma unroll
  for (int mf = 0; mf < 4; ++mf) {
#pragma unroll
    for (int nf = 0; nf < 2; ++nf) {
      int col = nt * BN + wn * 32 + nf * 16 + l16;
      v8f a = acc[mf][nf];
#pragma unroll
      for (int v = 0; v < 8; ++v) {
        int lrow = mt * BM + wm * 64 + mf * 16 + v + hi * 8;
        if (lrow < cnt)
          out[(size_t)(tok_start + lrow) * H_ + col] = a[v];
      }
    }
  }
}

// ---------------------------------------------------------------------------
extern "C" void kernel_launch(void* const* d_in, const int* in_sizes, int n_in,
                              void* d_out, int out_size, void* d_ws, size_t ws_size,
                              hipStream_t stream) {
  const float* x    = (const float*)d_in[0];
  const float* gate = (const float*)d_in[1];
  const float* up   = (const float*)d_in[2];
  const float* down = (const float*)d_in[3];
  const int*   tpe  = (const int*)d_in[4];
  float* out = (float*)d_out;
  unsigned short* inter = (unsigned short*)d_ws;   // [T, I] bf16 (46.1 MB)

  dim3 blk(256);
  dim3 g1(I_ / BN, T_ / BM, E_);   // (22, 64, 8)
  dim3 g2(H_ / BN, T_ / BM, E_);   // ( 8, 64, 8)
  moe_gateup_swiglu<<<g1, blk, 0, stream>>>(x, gate, up, tpe, inter);
  moe_down<<<g2, blk, 0, stream>>>(inter, down, tpe, out);
}
